// PointCloudTokenizer_54047868453420
// MI455X (gfx1250) — matrix operand order repaired and checked by reading, not compiled, not run on hardware
//
#include <hip/hip_runtime.h>

// ---------------------------------------------------------------------------
// Types for CDNA5 WMMA (wave32)
// ---------------------------------------------------------------------------
typedef __bf16 bf16;
typedef __attribute__((ext_vector_type(16))) __bf16 v16bf;
typedef __attribute__((ext_vector_type(8)))  __bf16 v8bf;
typedef __attribute__((ext_vector_type(8)))  float  v8f;

// Problem constants (match reference)
#define B_EV   8
#define NPTS   16384
#define FEAT_D 32
#define MTOK   128
#define KNN_K  16
#define DTOK   768
#define HID1   256
#define HID2   512
#define HID3   768
#define P_TOT  (B_EV * NPTS)      // 131072
#define ROWS_PER_WG 64            // rows per workgroup in MLP kernels

#define WMMA_BF16(a, b, c) \
    __builtin_amdgcn_wmma_f32_16x16x32_bf16(false, (a), false, (b), (short)0, (c), false, false)

// single-instruction ReLU: v_med3_f32(x, 0, +inf)
__device__ __forceinline__ float relu1(float x) {
    return __builtin_amdgcn_fmed3f(x, 0.0f, __builtin_inff());
}

// ---------------------------------------------------------------------------
// WMMA fragment loaders (bf16 16x16x32, contiguous K order, same for A and B)
//   A (16x32): lane L -> row = L%16 ; half = L/16 ;
//              elems 0..15 -> K = k0 + half*16 + 0..15
//   B (32x16): lane L -> col = L%16 ; elems 0..15 -> K = k0 + (L/16)*16 + 0..15
// ---------------------------------------------------------------------------
__device__ __forceinline__ v16bf load_a_frag(const bf16* __restrict__ act,
                                             int row_base, int ldk, int k0) {
    const int lane = threadIdx.x & 31;
    const int half = lane >> 4;
    const int r    = lane & 15;
    const bf16* p  = act + (row_base + r) * ldk + k0 + half * 16;
    v8bf lo = *(const v8bf*)(p);        // 16B LDS load
    v8bf hi = *(const v8bf*)(p + 8);    // 16B LDS load
    return __builtin_shufflevector(lo, hi, 0,1,2,3,4,5,6,7,8,9,10,11,12,13,14,15);
}

__device__ __forceinline__ v16bf load_b_frag(const bf16* __restrict__ wt,  // [N, K]
                                             int n0, int Kdim, int k0) {
    const int lane = threadIdx.x & 31;
    const int half = lane >> 4;
    const int n    = n0 + (lane & 15);
    const bf16* p  = wt + (size_t)n * Kdim + k0 + half * 16;
    return *(const v16bf*)(p);          // 32B global load (2 x b128)
}

// One MLP layer on a 64-row tile resident in LDS.
// 8 waves: wave w owns column tiles w, w+8, ... and computes ALL 4 row tiles
// against a single B fragment (each weight fragment loaded once per WG, and
// 4 independent WMMAs per B load hide the WMMA result hazard).
// Bias is column-only -> folded into the accumulator initializer (WMMA C-in).
// MODE: 0 = write bf16 to LDS, 1 = write bf16 to global, 2 = write f32 to global
template <int MODE, bool RELU>
__device__ __forceinline__ void mlp_layer(const bf16* __restrict__ in,   // LDS [64, Kdim]
                                          void* __restrict__ outp,      // stride Nout
                                          const bf16* __restrict__ wt,  // [Nout, Kdim]
                                          const float* __restrict__ bias,
                                          int Kdim, int Nout) {
    const int w      = threadIdx.x >> 5;
    const int lane   = threadIdx.x & 31;
    const int half   = lane >> 4;
    const int c      = lane & 15;
    const int ntiles = Nout >> 4;
    for (int nt = w; nt < ntiles; nt += 8) {
        const int n0 = nt << 4;
        const float bv = bias[n0 + c];    // this lane's output column bias
        v8f acc0 = {bv, bv, bv, bv, bv, bv, bv, bv};
        v8f acc1 = acc0, acc2 = acc0, acc3 = acc0;
        for (int k0 = 0; k0 < Kdim; k0 += 32) {
            const v16bf b  = load_b_frag(wt, n0, Kdim, k0);
            const v16bf a0 = load_a_frag(in,  0, Kdim, k0);
            const v16bf a1 = load_a_frag(in, 16, Kdim, k0);
            const v16bf a2 = load_a_frag(in, 32, Kdim, k0);
            const v16bf a3 = load_a_frag(in, 48, Kdim, k0);
            acc0 = WMMA_BF16(a0, b, acc0);
            acc1 = WMMA_BF16(a1, b, acc1);
            acc2 = WMMA_BF16(a2, b, acc2);
            acc3 = WMMA_BF16(a3, b, acc3);
        }
        v8f accs[4] = {acc0, acc1, acc2, acc3};
#pragma unroll
        for (int q = 0; q < 4; ++q) {
#pragma unroll
            for (int v = 0; v < 8; ++v) {
                float x = accs[q][v];
                if (RELU) x = relu1(x);
                const int row = q * 16 + v + half * 8;   // C/D: VGPR v -> M = v + 8*half
                const size_t o = (size_t)row * Nout + n0 + c;
                if (MODE == 0)      ((bf16*)outp)[o]  = (bf16)x;
                else if (MODE == 1) ((bf16*)outp)[o]  = (bf16)x;
                else                ((float*)outp)[o] = x;
            }
        }
    }
}

// ---------------------------------------------------------------------------
// Kernel: convert fp32 weight [K,N] -> bf16 transposed [N,K]
// ---------------------------------------------------------------------------
__global__ void __launch_bounds__(256) wt_transpose_bf16(const float* __restrict__ W,
                                                         bf16* __restrict__ Wt,
                                                         int Kd, int Nd) {
    int idx = blockIdx.x * 256 + threadIdx.x;
    if (idx < Kd * Nd) {
        int k = idx / Nd, n = idx - k * Nd;
        Wt[(size_t)n * Kd + k] = (bf16)W[idx];
    }
}

// ---------------------------------------------------------------------------
// Kernel: fused point MLP  32 -> 256 -> 512 -> 768 -> 768 (bf16 WMMA, f32 acc)
// dyn LDS: R1 = 64*512 bf16 (64KB), R2 = 64*768 bf16 (96KB)  -> 160KB total
// ---------------------------------------------------------------------------
__global__ void __launch_bounds__(256) point_mlp_kernel(
        const float* __restrict__ features,          // [P, 32]
        const bf16* __restrict__ wt1, const float* __restrict__ b1,
        const bf16* __restrict__ wt2, const float* __restrict__ b2,
        const bf16* __restrict__ wt3, const float* __restrict__ b3,
        const bf16* __restrict__ wt4, const float* __restrict__ b4,
        bf16* __restrict__ point_feats) {            // [P, 768] bf16
    extern __shared__ char smem[];
    bf16* R1 = (bf16*)smem;                          // cap 64*512
    bf16* R2 = (bf16*)(smem + ROWS_PER_WG * HID2 * 2);
    const int rowBase = blockIdx.x * ROWS_PER_WG;

    // stage x0 (64 x 32) fp32 -> bf16 into R1
    for (int i = threadIdx.x; i < ROWS_PER_WG * FEAT_D; i += 256)
        R1[i] = (bf16)features[(size_t)rowBase * FEAT_D + i];
    __syncthreads();

    mlp_layer<0, true>(R1, R2, wt1, b1, FEAT_D, HID1);   // x0 -> x1 [64,256] in R2
    __syncthreads();
    mlp_layer<0, true>(R2, R1, wt2, b2, HID1, HID2);     // x1 -> x2 [64,512] in R1
    __syncthreads();
    mlp_layer<0, true>(R1, R2, wt3, b3, HID2, HID3);     // x2 -> x3 [64,768] in R2
    __syncthreads();
    mlp_layer<1, false>(R2, point_feats + (size_t)rowBase * DTOK,
                        wt4, b4, HID3, DTOK);            // x3 -> global bf16
}

// ---------------------------------------------------------------------------
// Kernel: farthest-point sampling, one 1024-thread block per batch.
// wave32 shuffle argmax (first-index tie-break) + single LDS stage:
// 2 barriers per serial round instead of 10.
// ---------------------------------------------------------------------------
__global__ void __launch_bounds__(1024) fps_kernel(
        const float* __restrict__ coords,     // [P, 5]  (bidx,x,y,z,t)
        float* __restrict__ centroids_ws,     // [B, M, 4]
        float* __restrict__ centroids_out,    // d_out region [B, M, 4]
        float* __restrict__ masks_out) {      // d_out region [B, M]
    const int b = blockIdx.x;
    const float* base = coords + (size_t)b * NPTS * 5;

    if (threadIdx.x < MTOK) masks_out[b * MTOK + threadIdx.x] = 1.0f;

    float px[16], py[16], pz[16], pw[16], mind[16];
#pragma unroll
    for (int i = 0; i < 16; ++i) {
        const float* r = base + (size_t)(threadIdx.x + i * 1024) * 5;
        px[i] = r[1]; py[i] = r[2]; pz[i] = r[3]; pw[i] = r[4];
        mind[i] = 3.402823466e38f;
    }

    __shared__ float s_wd[32];
    __shared__ int   s_wi[32];
    __shared__ int   s_last;
    if (threadIdx.x == 0) s_last = 0;
    __syncthreads();

    for (int m = 0; m < MTOK; ++m) {
        const int last = s_last;
        // broadcast centroid via uniform global load (hits L0/L2, same addr all lanes)
        const float* cr = base + (size_t)last * 5;
        const float cx = cr[1], cy = cr[2], cz = cr[3], ct = cr[4];
        if (threadIdx.x == 0) {
            const size_t o = ((size_t)b * MTOK + m) * 4;
            centroids_ws[o + 0] = cx; centroids_ws[o + 1] = cy;
            centroids_ws[o + 2] = cz; centroids_ws[o + 3] = ct;
            centroids_out[o + 0] = cx; centroids_out[o + 1] = cy;
            centroids_out[o + 2] = cz; centroids_out[o + 3] = ct;
        }
        float bestd = -1.f; int besti = 0x7fffffff;
#pragma unroll
        for (int i = 0; i < 16; ++i) {
            const float dx = px[i] - cx, dy = py[i] - cy,
                        dz = pz[i] - cz, dt = pw[i] - ct;
            const float d = dx * dx + dy * dy + dz * dz + dt * dt;
            mind[i] = fminf(mind[i], d);
            if (mind[i] > bestd) { bestd = mind[i]; besti = threadIdx.x + i * 1024; }
        }
        // wave32 argmax, first-index tie-break
#pragma unroll
        for (int off = 16; off > 0; off >>= 1) {
            const float od = __shfl_xor(bestd, off, 32);
            const int   oi = __shfl_xor(besti, off, 32);
            if (od > bestd || (od == bestd && oi < besti)) { bestd = od; besti = oi; }
        }
        if ((threadIdx.x & 31) == 0) {
            s_wd[threadIdx.x >> 5] = bestd;
            s_wi[threadIdx.x >> 5] = besti;
        }
        __syncthreads();
        if (threadIdx.x < 32) {
            float vd = s_wd[threadIdx.x];
            int   vi = s_wi[threadIdx.x];
#pragma unroll
            for (int off = 16; off > 0; off >>= 1) {
                const float od = __shfl_xor(vd, off, 32);
                const int   oi = __shfl_xor(vi, off, 32);
                if (od > vd || (od == vd && oi < vi)) { vd = od; vi = oi; }
            }
            if (threadIdx.x == 0) s_last = vi;
        }
        __syncthreads();
    }
}

// ---------------------------------------------------------------------------
// Kernel: KNN(16) + gather + max-pool. One 256-thread WG per (b, m).
// Packed key (d2_bits<<32 | idx) is order-preserving since d2 >= 0.
// ---------------------------------------------------------------------------
__global__ void __launch_bounds__(256) knn_pool_kernel(
        const float* __restrict__ coords,      // [P, 5]
        const float* __restrict__ centroids,   // [B, M, 4]
        const bf16*  __restrict__ point_feats, // [P, 768]
        bf16* __restrict__ pooled) {           // [B*M, 768]
    const int bm = blockIdx.x;
    const int b  = bm >> 7;                    // / MTOK
    const float* cen = centroids + (size_t)bm * 4;
    const float cx = cen[0], cy = cen[1], cz = cen[2], ct = cen[3];
    const float* base = coords + (size_t)b * NPTS * 5;

    unsigned long long heap[KNN_K];
#pragma unroll
    for (int j = 0; j < KNN_K; ++j) heap[j] = 0xFFFFFFFFFFFFFFFFull;

    for (int p = threadIdx.x; p < NPTS; p += 256) {
        const float* r = base + (size_t)p * 5;
        const float dx = r[1] - cx, dy = r[2] - cy, dz = r[3] - cz, dt = r[4] - ct;
        const float d = dx * dx + dy * dy + dz * dz + dt * dt;
        unsigned long long key =
            ((unsigned long long)__float_as_uint(d) << 32) | (unsigned)p;
        if (key < heap[KNN_K - 1]) {
#pragma unroll
            for (int j = 0; j < KNN_K; ++j) {      // sorted-insert compare-swap chain
                const unsigned long long lo = key < heap[j] ? key : heap[j];
                const unsigned long long hi = key < heap[j] ? heap[j] : key;
                heap[j] = lo; key = hi;
            }
        }
    }

    __shared__ unsigned long long allk[256 * KNN_K];
    __shared__ unsigned long long s_wmin[8];
    __shared__ int s_sel[KNN_K];
#pragma unroll
    for (int j = 0; j < KNN_K; ++j) allk[threadIdx.x * KNN_K + j] = heap[j];
    __syncthreads();

    int head = 0;
    for (int k = 0; k < KNN_K; ++k) {
        const unsigned long long mine = allk[threadIdx.x * KNN_K + head];
        unsigned long long v = mine;
#pragma unroll
        for (int off = 16; off > 0; off >>= 1) {
            const unsigned long long o = __shfl_xor(v, off, 32);
            v = (o < v) ? o : v;
        }
        if ((threadIdx.x & 31) == 0) s_wmin[threadIdx.x >> 5] = v;
        __syncthreads();
        unsigned long long win = s_wmin[0];
#pragma unroll
        for (int j = 1; j < 8; ++j) win = (s_wmin[j] < win) ? s_wmin[j] : win;
        if (mine == win) {                        // unique keys: exactly one winner
            s_sel[k] = (int)(win & 0xFFFFFFFFull);
            ++head;
        }
        __syncthreads();
    }

    // gather 16 bf16 feature rows, max-pool over K
    const size_t fbase = (size_t)b * NPTS;
    for (int dcol = threadIdx.x; dcol < DTOK; dcol += 256) {
        float mx = -3.402823466e38f;
#pragma unroll
        for (int k = 0; k < KNN_K; ++k) {
            const float v =
                (float)point_feats[(fbase + (unsigned)s_sel[k]) * DTOK + dcol];
            mx = fmaxf(mx, v);
        }
        pooled[(size_t)bm * DTOK + dcol] = (bf16)mx;
    }
}

// ---------------------------------------------------------------------------
// Kernel: token MLP  768 -> ReLU -> 768 (WMMA), fp32 tokens to d_out
// dyn LDS: 2 x 64*768 bf16 = 192KB
// ---------------------------------------------------------------------------
__global__ void __launch_bounds__(256) token_mlp_kernel(
        const bf16* __restrict__ pooled,      // [1024, 768]
        const bf16* __restrict__ wtn1, const float* __restrict__ bn1,
        const bf16* __restrict__ wtn2, const float* __restrict__ bn2,
        float* __restrict__ tokens_out) {     // [1024, 768]
    extern __shared__ char smem[];
    bf16* R1 = (bf16*)smem;
    bf16* R2 = (bf16*)(smem + ROWS_PER_WG * DTOK * 2);
    const int rowBase = blockIdx.x * ROWS_PER_WG;

    for (int i = threadIdx.x; i < ROWS_PER_WG * DTOK; i += 256)
        R1[i] = pooled[(size_t)rowBase * DTOK + i];
    __syncthreads();

    mlp_layer<0, true>(R1, R2, wtn1, bn1, DTOK, DTOK);
    __syncthreads();
    mlp_layer<2, false>(R2, tokens_out + (size_t)rowBase * DTOK,
                        wtn2, bn2, DTOK, DTOK);
}

// ---------------------------------------------------------------------------
// Host launcher
// ---------------------------------------------------------------------------
extern "C" void kernel_launch(void* const* d_in, const int* in_sizes, int n_in,
                              void* d_out, int out_size, void* d_ws, size_t ws_size,
                              hipStream_t stream) {
    (void)in_sizes; (void)n_in; (void)out_size; (void)ws_size;

    const float* coords   = (const float*)d_in[0];
    const float* features = (const float*)d_in[1];
    const float* W1 = (const float*)d_in[3];  const float* b1 = (const float*)d_in[4];
    const float* W2 = (const float*)d_in[5];  const float* b2 = (const float*)d_in[6];
    const float* W3 = (const float*)d_in[7];  const float* b3 = (const float*)d_in[8];
    const float* W4 = (const float*)d_in[9];  const float* b4 = (const float*)d_in[10];
    const float* Wn1 = (const float*)d_in[11]; const float* bn1 = (const float*)d_in[12];
    const float* Wn2 = (const float*)d_in[13]; const float* bn2 = (const float*)d_in[14];

    // workspace bump allocator (256B aligned)
    char* ws = (char*)d_ws;
    size_t off = 0;
    auto alloc = [&](size_t bytes) -> char* {
        char* p = ws + off;
        off += (bytes + 255) & ~(size_t)255;
        return p;
    };
    bf16* wt1  = (bf16*)alloc((size_t)FEAT_D * HID1 * 2);
    bf16* wt2  = (bf16*)alloc((size_t)HID1   * HID2 * 2);
    bf16* wt3  = (bf16*)alloc((size_t)HID2   * HID3 * 2);
    bf16* wt4  = (bf16*)alloc((size_t)HID3   * DTOK * 2);
    bf16* wtn1 = (bf16*)alloc((size_t)DTOK   * DTOK * 2);
    bf16* wtn2 = (bf16*)alloc((size_t)DTOK   * DTOK * 2);
    bf16* point_feats = (bf16*)alloc((size_t)P_TOT * DTOK * 2);
    float* centroids_ws = (float*)alloc((size_t)B_EV * MTOK * 4 * 4);
    bf16* pooled = (bf16*)alloc((size_t)B_EV * MTOK * DTOK * 2);

    float* tokens_out    = (float*)d_out;                    // [8,128,768]
    float* centroids_out = tokens_out + (size_t)B_EV * MTOK * DTOK;
    float* masks_out     = centroids_out + (size_t)B_EV * MTOK * 4;

    auto tr = [&](const float* W, bf16* Wt, int Kd, int Nd) {
        const int total = Kd * Nd;
        wt_transpose_bf16<<<(total + 255) / 256, 256, 0, stream>>>(W, Wt, Kd, Nd);
    };
    tr(W1, wt1, FEAT_D, HID1);
    tr(W2, wt2, HID1, HID2);
    tr(W3, wt3, HID2, HID3);
    tr(W4, wt4, HID3, DTOK);
    tr(Wn1, wtn1, DTOK, DTOK);
    tr(Wn2, wtn2, DTOK, DTOK);

    point_mlp_kernel<<<P_TOT / ROWS_PER_WG, 256,
                       (ROWS_PER_WG * HID2 + ROWS_PER_WG * HID3) * 2, stream>>>(
        features, wt1, b1, wt2, b2, wt3, b3, wt4, b4, point_feats);

    fps_kernel<<<B_EV, 1024, 0, stream>>>(coords, centroids_ws, centroids_out,
                                          masks_out);

    knn_pool_kernel<<<B_EV * MTOK, 256, 0, stream>>>(coords, centroids_ws,
                                                     point_feats, pooled);

    token_mlp_kernel<<<(B_EV * MTOK) / ROWS_PER_WG, 256,
                       2 * ROWS_PER_WG * DTOK * 2, stream>>>(
        pooled, wtn1, bn1, wtn2, bn2, tokens_out);
}
